// RengarNetwork_67422396612990
// MI455X (gfx1250) — compile-verified, matrix operand order checked by reading, not run on hardware
//
#include <hip/hip_runtime.h>
#include <stdint.h>

// ---------------- problem constants ----------------
#define VOCAB   185
#define NK      6        // K tiles of 32 (vocab padded 185 -> 192)
#define NT1     48       // stage-1 N tiles of 16 (768 feature cols)
#define LFEAT   32
#define THRESH  (127.0f/128.0f)

// ---------------- workspace layout (bytes) ----------------
// wsB: stage-1 table, bf16, pre-swizzled into WMMA B-fragment order.
//      frag(kt,nt) = 32x16 bf16 = 1024B; cols 0..255=emb_fs, 256..511=emb_va,
//      512..639=emb_ha, 640..767=emb_ra; rows(K)=vocab idx (>=185 zeroed).
#define WSB_BYTES   (NK*NT1*1024)          // 294912
// wsW: stage-2 weights as B-fragments. frags 0..15: [fs_w | absva_w] col-pairs,
//      16..23: [absha_w | absra_w], 24..39: [va_w | fsxva_w], 40..43: haxra_w.
#define NFRAG_W     44
#define WSW_BYTES   (NFRAG_W*1024)

// ---------------- per-wave LDS layout ----------------
#define CNT_ROWSTRIDE 400                  // 192 bf16 + pad (bank spread)
#define CNT_MAT   (16*CNT_ROWSTRIDE)       // one counts matrix, 16 samples
#define CNT_BYTES (4*CNT_MAT)              // 25600
#define AST_OFF   CNT_BYTES
#define AST_ROWSTRIDE 80                   // 16x32 bf16 staging + pad
#define AST_BYTES (16*AST_ROWSTRIDE)       // 1280
#define WTM_OFF   (AST_OFF+AST_BYTES)
#define PER_WAVE  27136
#define WPB       8                        // waves per block (256 threads)
#define LDS_TOTAL (PER_WAVE*WPB)           // 217088 <= 320KB WGP LDS

typedef __attribute__((ext_vector_type(16))) __bf16 v16bf;
typedef __attribute__((ext_vector_type(8)))  float  v8f;

union FragU { uint4 q[2]; v16bf v; };

__device__ __forceinline__ uint16_t f2bf(float f) {
  uint32_t u = __float_as_uint(f);
  return (uint16_t)((u + 0x7FFFu + ((u >> 16) & 1u)) >> 16);   // RNE
}
__device__ __forceinline__ float bf2f(uint16_t h) {
  return __uint_as_float(((uint32_t)h) << 16);
}
__device__ __forceinline__ float clampT(float x) {
  return fminf(fmaxf(x, -THRESH), THRESH);
}
__device__ __forceinline__ void wave_lds_fence() {
  __builtin_amdgcn_wave_barrier();
  asm volatile("s_wait_dscnt 0x0" ::: "memory");
  __builtin_amdgcn_wave_barrier();
}
__device__ __forceinline__ v8f wmma_bf16(v16bf a, v16bf b, v8f c) {
  return __builtin_amdgcn_wmma_f32_16x16x32_bf16(false, a, false, b,
                                                 (short)0, c, false, false);
}
// B-fragment: lane*32 bytes contiguous -> two global_load_b128 per lane.
__device__ __forceinline__ v16bf load_frag_global(const uint16_t* ws, int frag, int lane) {
  const char* p = (const char*)ws + frag * 1024 + lane * 32;
  FragU u; u.q[0] = *(const uint4*)p; u.q[1] = *(const uint4*)(p + 16);
  return u.v;
}
// A-fragment from row-major bf16 counts in LDS (lane m: K hi*8+0..7 and 16+hi*8+0..7).
__device__ __forceinline__ v16bf load_counts_frag(const char* mat, int kt, int lane) {
  const int m = lane & 15, hi = lane >> 4;
  const char* p = mat + m * CNT_ROWSTRIDE + kt * 64 + hi * 16;
  FragU u; u.q[0] = *(const uint4*)p; u.q[1] = *(const uint4*)(p + 32);
  return u.v;
}
__device__ __forceinline__ v16bf load_astage(const char* ast, int lane) {
  const int m = lane & 15, hi = lane >> 4;
  const char* p = ast + m * AST_ROWSTRIDE + hi * 16;
  FragU u; u.q[0] = *(const uint4*)p; u.q[1] = *(const uint4*)(p + 32);
  return u.v;
}
__device__ __forceinline__ void bf16_rmw_add(uint16_t* p, float v) {
  *p = f2bf(bf2f(*p) + v);
}
// Store one 16x16 f32 C-layout tile into Astage cols [c0..c0+15] as bf16.
__device__ __forceinline__ void store_tile(char* ast, int c0, const float* t, int lane) {
  const int n = lane & 15, hi = lane >> 4;
#pragma unroll
  for (int r = 0; r < 8; r++)
    ((uint16_t*)(ast + (r + hi * 8) * AST_ROWSTRIDE))[c0 + n] = f2bf(t[r]);
}
// In-lane K index for a B-fragment element (lane, half j).
__device__ __forceinline__ int frag_kk(int lane, int j) {
  return ((lane >> 4) << 3) + (j & 7) + ((j >> 3) << 4);
}

// ================= prep kernel 1: swizzle embedding tables to bf16 B-frags =================
__global__ void rengar_prep_tables(const float* __restrict__ emb_fs,
                                   const float* __restrict__ emb_va,
                                   const float* __restrict__ emb_ha,
                                   const float* __restrict__ emb_ra,
                                   uint16_t* __restrict__ wsB) {
  const int tid = blockIdx.x * blockDim.x + threadIdx.x;
  if (tid >= NK * NT1 * 512) return;
  const int frag = tid >> 9, within = tid & 511;
  const int lane = within >> 4, j = within & 15;
  const int kt = frag / NT1, nt = frag % NT1;
  const int k = kt * 32 + frag_kk(lane, j);
  const int ng = nt * 16 + (lane & 15);
  float v = 0.0f;
  if (k < VOCAB) {
    if (ng < 256)      v = emb_fs[k * 256 + ng];
    else if (ng < 512) v = emb_va[k * 256 + (ng - 256)];
    else if (ng < 640) v = emb_ha[k * 128 + (ng - 512)];
    else               v = emb_ra[k * 128 + (ng - 640)];
  }
  wsB[frag * 512 + within] = f2bf(v);
}

// ================= prep kernel 2: swizzle stage-2 weights to bf16 B-frags =================
__global__ void rengar_prep_weights(const float* __restrict__ fs_w,
                                    const float* __restrict__ absva_w,
                                    const float* __restrict__ absha_w,
                                    const float* __restrict__ absra_w,
                                    const float* __restrict__ va_w,
                                    const float* __restrict__ fsxva_w,
                                    const float* __restrict__ haxra_w,
                                    uint16_t* __restrict__ wsW) {
  const int tid = blockIdx.x * blockDim.x + threadIdx.x;
  if (tid >= NFRAG_W * 512) return;
  const int f = tid >> 9, within = tid & 511;
  const int lane = within >> 4, j = within & 15;
  const int kk = frag_kk(lane, j);
  const int nn = lane & 15;                      // output feature (S3=16)
  float v = 0.0f;
  if (f < 24) {                                  // fs2 path (X = [fs1,|va1|,|ha1|,|ra1|])
    if (f < 16) {
      const int c = f * 16 + (kk & 15);
      v = (kk < 16) ? fs_w[nn * 256 + c] : absva_w[nn * 256 + c];
    } else {
      const int c = (f - 16) * 16 + (kk & 15);
      v = (kk < 16) ? absha_w[nn * 128 + c] : absra_w[nn * 128 + c];
    }
  } else {                                       // va2 path (X = [va1,fs1*va1,ha1*ra1])
    const int g = f - 24;
    if (g < 16) {
      const int c = g * 16 + (kk & 15);
      v = (kk < 16) ? va_w[nn * 256 + c] : fsxva_w[nn * 256 + c];
    } else {
      v = haxra_w[nn * 128 + (g - 16) * 32 + kk];
    }
  }
  wsW[f * 512 + within] = f2bf(v);
}

// ================= main fused kernel: one wave = 16 samples =================
__global__ __launch_bounds__(256, 1)
void rengar_main(const int* __restrict__ pst_idx,
                 const float* __restrict__ color_sign,
                 const float* __restrict__ sob_sign,
                 const float* __restrict__ wtm,
                 const float* __restrict__ tempo_w,
                 const float* __restrict__ fs_b,
                 const float* __restrict__ out_va_w,
                 const float* __restrict__ out_fsxva_w,
                 const uint16_t* __restrict__ wsB,
                 const uint16_t* __restrict__ wsW,
                 float* __restrict__ out) {
  extern __shared__ char smem[];
  const int lane = threadIdx.x & 31;
  const int wave = threadIdx.x >> 5;
  char* L = smem + wave * PER_WAVE;
  char* ast = L + AST_OFF;
  const int base = (blockIdx.x * WPB + wave) * 16;
  const int n = lane & 15;
  const int hi = lane >> 4;
  const v8f vzero = {0.f, 0.f, 0.f, 0.f, 0.f, 0.f, 0.f, 0.f};

  // ---- zero counts (rows >= VOCAB stay zero -> K padding to 192) ----
  for (int off = lane * 16; off < CNT_BYTES; off += 32 * 16) {
    uint4 z; z.x = z.y = z.z = z.w = 0u;
    *(uint4*)(L + off) = z;
  }
  if (lane < 16) ((float*)(L + WTM_OFF))[lane] = wtm[base + lane];
  wave_lds_fence();

  // ---- scatter per-sample bag weights into counts (lane = sample, halves split work) ----
  {
    const int s = n;
    const int* ip = pst_idx   + (base + s) * LFEAT;
    const float* cp = color_sign + (base + s) * LFEAT;
    const float* sp = sob_sign   + (base + s) * LFEAT;
    uint16_t* m0 = (uint16_t*)(L + 0 * CNT_MAT + s * CNT_ROWSTRIDE);
    uint16_t* m1 = (uint16_t*)(L + 1 * CNT_MAT + s * CNT_ROWSTRIDE);
    uint16_t* m2 = (uint16_t*)(L + 2 * CNT_MAT + s * CNT_ROWSTRIDE);
    uint16_t* m3 = (uint16_t*)(L + 3 * CNT_MAT + s * CNT_ROWSTRIDE);
    for (int i = 0; i < LFEAT; i++) {
      const int idx = ip[i];
      const float c = cp[i];
      if (hi == 0) {
        bf16_rmw_add(m0 + idx, 1.0f);
        bf16_rmw_add(m1 + idx, c);
      } else {
        const float sv = sp[i];
        bf16_rmw_add(m2 + idx, sv);
        bf16_rmw_add(m3 + idx, c * sv);
      }
    }
  }
  wave_lds_fence();

  const float* wtmv = (const float*)(L + WTM_OFF);
  v8f accFS2 = vzero, accVA2 = vzero;
  float prevP[8];

  // ---- S1 sections (fs & va, 16 N-tiles each), stage-2 fused ----
  for (int j = 0; j < 16; j++) {
    v8f aF = vzero, aV = vzero;
#pragma unroll
    for (int kt = 0; kt < NK; kt++) {
      v16bf cF = load_counts_frag(L + 0 * CNT_MAT, kt, lane);
      v16bf bF = load_frag_global(wsB, kt * NT1 + j, lane);
      v16bf cV = load_counts_frag(L + 1 * CNT_MAT, kt, lane);
      v16bf bV = load_frag_global(wsB, kt * NT1 + 16 + j, lane);
      aF = wmma_bf16(cF, bF, aF);
      aV = wmma_bf16(cV, bV, aV);
    }
    const float tv = tempo_w[j * 16 + n];
    float fs1v[8], va1v[8], absv[8], prod[8];
#pragma unroll
    for (int r = 0; r < 8; r++) {
      fs1v[r] = clampT(aF[r]);
      va1v[r] = clampT(aV[r] + wtmv[r + hi * 8] * tv);
      absv[r] = fabsf(va1v[r]);
      prod[r] = fs1v[r] * va1v[r];
    }
    store_tile(ast, 0, fs1v, lane);             // fs2 path: [fs1 | |va1|]
    store_tile(ast, 16, absv, lane);
    wave_lds_fence();
    accFS2 = wmma_bf16(load_astage(ast, lane), load_frag_global(wsW, j, lane), accFS2);
    wave_lds_fence();
    store_tile(ast, 0, va1v, lane);             // va2 path: [va1 | fs1*va1]
    store_tile(ast, 16, prod, lane);
    wave_lds_fence();
    accVA2 = wmma_bf16(load_astage(ast, lane), load_frag_global(wsW, 24 + j, lane), accVA2);
    wave_lds_fence();
  }

  // ---- S2 sections (ha & ra, 8 N-tiles each) ----
  for (int j = 0; j < 8; j++) {
    v8f aH = vzero, aR = vzero;
#pragma unroll
    for (int kt = 0; kt < NK; kt++) {
      v16bf cH = load_counts_frag(L + 2 * CNT_MAT, kt, lane);
      v16bf bH = load_frag_global(wsB, kt * NT1 + 32 + j, lane);
      v16bf cR = load_counts_frag(L + 3 * CNT_MAT, kt, lane);
      v16bf bR = load_frag_global(wsB, kt * NT1 + 40 + j, lane);
      aH = wmma_bf16(cH, bH, aH);
      aR = wmma_bf16(cR, bR, aR);
    }
    float hav[8], rav[8], ah[8], ar[8], pr[8];
#pragma unroll
    for (int r = 0; r < 8; r++) {
      hav[r] = clampT(aH[r]); rav[r] = clampT(aR[r]);
      ah[r] = fabsf(hav[r]);  ar[r] = fabsf(rav[r]);
      pr[r] = hav[r] * rav[r];
    }
    store_tile(ast, 0, ah, lane);               // fs2 path: [|ha1| | |ra1|]
    store_tile(ast, 16, ar, lane);
    wave_lds_fence();
    accFS2 = wmma_bf16(load_astage(ast, lane), load_frag_global(wsW, 16 + j, lane), accFS2);
    wave_lds_fence();
    if ((j & 1) == 0) {
#pragma unroll
      for (int r = 0; r < 8; r++) prevP[r] = pr[r];   // stash ha*ra tile, pair with next
    } else {
      store_tile(ast, 0, prevP, lane);          // va2 path: [p_{j-1} | p_j]
      store_tile(ast, 16, pr, lane);
      wave_lds_fence();
      accVA2 = wmma_bf16(load_astage(ast, lane),
                         load_frag_global(wsW, 40 + (j >> 1), lane), accVA2);
      wave_lds_fence();
    }
  }

  // ---- output layer + 16-lane feature reduction ----
  const float bn  = fs_b[n];
  const float wov = out_va_w[n];
  const float wof = out_fsxva_w[n];
#pragma unroll
  for (int r = 0; r < 8; r++) {
    const float fs2 = clampT(accFS2[r] + bn);
    const float va2 = clampT(accVA2[r]);
    float v = va2 * wov + fs2 * va2 * wof;
    v += __shfl_xor(v, 1, 32);
    v += __shfl_xor(v, 2, 32);
    v += __shfl_xor(v, 4, 32);
    v += __shfl_xor(v, 8, 32);
    if (n == 0) out[base + r + hi * 8] = v;     // lane0 -> row r, lane16 -> row r+8
  }
}

// ================= host launcher =================
extern "C" void kernel_launch(void* const* d_in, const int* in_sizes, int n_in,
                              void* d_out, int out_size, void* d_ws, size_t ws_size,
                              hipStream_t stream) {
  (void)n_in; (void)out_size; (void)ws_size;
  const int*   pst_idx   = (const int*)  d_in[0];
  const float* color     = (const float*)d_in[1];
  const float* sob       = (const float*)d_in[2];
  const float* wtm       = (const float*)d_in[3];
  const float* emb_fs    = (const float*)d_in[4];
  const float* emb_va    = (const float*)d_in[5];
  const float* emb_ha    = (const float*)d_in[6];
  const float* emb_ra    = (const float*)d_in[7];
  const float* tempo_w   = (const float*)d_in[8];
  const float* fs_w      = (const float*)d_in[9];
  const float* fs_b      = (const float*)d_in[10];
  const float* absva_w   = (const float*)d_in[11];
  const float* absha_w   = (const float*)d_in[12];
  const float* absra_w   = (const float*)d_in[13];
  const float* va_w      = (const float*)d_in[14];
  const float* fsxva_w   = (const float*)d_in[15];
  const float* haxra_w   = (const float*)d_in[16];
  const float* out_va_w  = (const float*)d_in[17];
  const float* out_fsxva = (const float*)d_in[18];
  float* out = (float*)d_out;
  uint16_t* wsB = (uint16_t*)d_ws;
  uint16_t* wsW = (uint16_t*)((char*)d_ws + WSB_BYTES);

  (void)hipFuncSetAttribute((const void*)rengar_main,
                            hipFuncAttributeMaxDynamicSharedMemorySize, LDS_TOTAL);

  rengar_prep_tables<<<(NK * NT1 * 512 + 255) / 256, 256, 0, stream>>>(
      emb_fs, emb_va, emb_ha, emb_ra, wsB);
  rengar_prep_weights<<<(NFRAG_W * 512 + 255) / 256, 256, 0, stream>>>(
      fs_w, absva_w, absha_w, absra_w, va_w, fsxva_w, haxra_w, wsW);

  const int B = in_sizes[3];                 // wtm: one entry per sample
  const int grid = B / (WPB * 16);           // 128 samples per 256-thread block
  rengar_main<<<grid, 256, LDS_TOTAL, stream>>>(
      pst_idx, color, sob, wtm, tempo_w, fs_b, out_va_w, out_fsxva,
      wsB, wsW, out);
}
